// CustomLSTM_80341658239359
// MI455X (gfx1250) — matrix operand order repaired and checked by reading, not compiled, not run on hardware
//
#include <hip/hip_runtime.h>
#include <hip/hip_bf16.h>
#include <stdint.h>

// ---------------- problem dimensions ----------------
#define S_LEN 512
#define BATCH 64
#define ISZ   512
#define HSZ   1024
#define OSZ   512
#define CSZ   1536   // ISZ + HSZ
#define G4H   4096   // 4*HSZ

// ---------------- vector types ----------------
typedef __attribute__((ext_vector_type(16))) __bf16        v16bf;
typedef __attribute__((ext_vector_type(8)))  __bf16        v8bf;
typedef __attribute__((ext_vector_type(8)))  float         v8f;
typedef __attribute__((ext_vector_type(4)))  float         f32x4;
typedef __attribute__((ext_vector_type(4)))  unsigned int  u32x4;
typedef __attribute__((ext_vector_type(4)))  unsigned short u16x4;
typedef __attribute__((ext_vector_type(4)))  int           i32x4;

// address-space qualified int4 for the async-LDS builtin
typedef __attribute__((address_space(1))) i32x4 as1_i32x4;
typedef __attribute__((address_space(3))) i32x4 as3_i32x4;

// ---------------- CDNA5 async global->LDS (guarded) ----------------
#if __has_builtin(__builtin_amdgcn_global_load_async_to_lds_b128) && \
    __has_builtin(__builtin_amdgcn_s_wait_asynccnt)
#define ASYNC_LDS 1
#define WAIT_ASYNC(n) __builtin_amdgcn_s_wait_asynccnt(n)
#else
#define ASYNC_LDS 0
#define WAIT_ASYNC(n) ((void)0)
#endif

__device__ __forceinline__ void stage16(const unsigned short* __restrict__ g,
                                        unsigned short* l) {
#if ASYNC_LDS
  __builtin_amdgcn_global_load_async_to_lds_b128(
      (as1_i32x4*)(unsigned long long)g,
      (as3_i32x4*)(unsigned int)(unsigned long long)l,
      0, 0);
#else
  *(u32x4*)l = *(const u32x4*)g;
#endif
}

// ---------------- helpers ----------------
__device__ __forceinline__ unsigned short f2bf(float f) {
  unsigned int u = __float_as_uint(f);
  unsigned int r = u + 0x7FFFu + ((u >> 16) & 1u);   // round to nearest even
  return (unsigned short)(r >> 16);
}

__device__ __forceinline__ v16bf mk16(v8bf lo, v8bf hi) {
  v16bf r;
#pragma unroll
  for (int i = 0; i < 8; ++i) { r[i] = lo[i]; r[i + 8] = hi[i]; }
  return r;
}

__device__ __forceinline__ v8f wmma_bf16(v16bf a, v16bf b, v8f c) {
  return __builtin_amdgcn_wmma_f32_16x16x32_bf16(false, a, false, b,
                                                 (short)0, c, false, false);
}

__device__ __forceinline__ float sigm(float x) {
  return 1.0f / (1.0f + __expf(-x));
}
__device__ __forceinline__ float tanh_fast(float x) {
  return 2.0f / (1.0f + __expf(-2.0f * x)) - 1.0f;
}

// ---------------- one-time prep ----------------
__global__ void prep_weights(const float* __restrict__ Wf, const float* __restrict__ Wi,
                             const float* __restrict__ Wo, const float* __restrict__ Wg,
                             const float* __restrict__ bf_, const float* __restrict__ bi_,
                             const float* __restrict__ bo_, const float* __restrict__ bg_,
                             const float* __restrict__ Wy,
                             unsigned short* __restrict__ wbf,
                             unsigned short* __restrict__ wybf,
                             float* __restrict__ ball) {
  long i = (long)blockIdx.x * blockDim.x + threadIdx.x;
  const long NW = (long)G4H * CSZ;   // 6291456
  const long NY = (long)OSZ * HSZ;   // 524288
  if (i < NW) {
    int n = (int)(i / CSZ), k = (int)(i % CSZ);
    int gate = n >> 10, row = n & 1023;
    const float* Wsel = (gate == 0) ? Wf : (gate == 1) ? Wi : (gate == 2) ? Wo : Wg;
    wbf[i] = f2bf(Wsel[(size_t)row * CSZ + k]);
  } else if (i < NW + NY) {
    long j = i - NW;
    wybf[j] = f2bf(Wy[j]);
  } else if (i < NW + NY + G4H) {
    int n = (int)(i - NW - NY);
    int gate = n >> 10, row = n & 1023;
    const float* bsel = (gate == 0) ? bf_ : (gate == 1) ? bi_ : (gate == 2) ? bo_ : bg_;
    ball[n] = bsel[row];
  }
}

__global__ void prep_x(const float* __restrict__ x, unsigned short* __restrict__ xbf) {
  long i = ((long)blockIdx.x * blockDim.x + threadIdx.x) * 4;
  f32x4 v = *(const f32x4*)(x + i);
  u16x4 o;
#pragma unroll
  for (int j = 0; j < 4; ++j) o[j] = f2bf(v[j]);
  *(u16x4*)(xbf + i) = o;
}

__global__ void init_state(unsigned short* __restrict__ hbf, float* __restrict__ c) {
  int i = blockIdx.x * blockDim.x + threadIdx.x;   // 65536 total
  hbf[i] = 0;
  c[i] = 0.0f;
}

// ---------------- per-step: gate GEMM [64 x 1536] x [1536 x 4096] ----------------
// Block: 256 threads = 8 waves covering M=64 x N=64 (wave tile 16x32).
// Grid: 64 blocks over N=4096.  B tiles (32K x 64N) staged through a 4-deep
// LDS ring via async global->LDS DMA, prefetch depth 3.
#define LDS_COL 40   // ushorts per column (32 data + 8 pad -> 80B stride)
#define KT      (CSZ / 32)   // 48 K-chunks
#define KTX     (ISZ / 32)   // 16 chunks from x
 
__global__ __launch_bounds__(256) void gates_wmma(
    const unsigned short* __restrict__ xbf,   // [S,B,ISZ]
    const unsigned short* __restrict__ hbf,   // [B,HSZ]
    const unsigned short* __restrict__ wbf,   // [G4H,CSZ]
    const float* __restrict__ ball,           // [G4H]
    float* __restrict__ preact,               // [B,G4H]
    int t) {
  __shared__ unsigned short ldsB[4][64 * LDS_COL];   // 20 KB ring

  const int tid  = threadIdx.x;
  const int wave = tid >> 5;
  const int lane = tid & 31;
  const int lr   = lane & 15;   // row/col within 16
  const int kh   = lane >> 4;   // K half select

  const int m0    = (wave & 3) * 16;
  const int nhalf = wave >> 2;            // 0..1 -> 32-col half
  const int nblk  = blockIdx.x * 64;

  // staging coordinates: 256 threads move 64 cols x 64B per K-chunk
  const int scol  = tid >> 2;
  const int spart = tid & 3;
  const unsigned short* gsrc = wbf + (size_t)(nblk + scol) * CSZ + spart * 8;
  const int soff = scol * LDS_COL + spart * 8;

  const int nbase = nblk + nhalf * 32 + lr;
  const float b0 = ball[nbase];
  const float b1 = ball[nbase + 16];
  v8f acc0, acc1;
#pragma unroll
  for (int i = 0; i < 8; ++i) { acc0[i] = b0; acc1[i] = b1; }

  const unsigned short* xrow = xbf + ((size_t)t * BATCH + (m0 + lr)) * ISZ;
  const unsigned short* hrow = hbf + (size_t)(m0 + lr) * HSZ;

  const int c0off = (nhalf * 32 + lr) * LDS_COL + kh * 16;
  const int c1off = c0off + 16 * LDS_COL;

  // prologue: prefetch chunks 0..2
  stage16(gsrc,      &ldsB[0][soff]);
  stage16(gsrc + 32, &ldsB[1][soff]);
  stage16(gsrc + 64, &ldsB[2][soff]);

  auto kstep = [&](const unsigned short* aptr, int kt) {
    const int nxt = kt + 3;
    if (nxt < KT) {
      stage16(gsrc + (size_t)nxt * 32, &ldsB[nxt & 3][soff]);
      WAIT_ASYNC(3);                     // chunk kt's DMA complete
    } else if (kt == KT - 3) {
      WAIT_ASYNC(2);
    } else if (kt == KT - 2) {
      WAIT_ASYNC(1);
    } else {
      WAIT_ASYNC(0);
    }
    __syncthreads();

    v16bf A = mk16(*(const v8bf*)(aptr + kh * 16),
                   *(const v8bf*)(aptr + kh * 16 + 8));

    const unsigned short* lbuf = ldsB[kt & 3];
    const unsigned short* c0p = lbuf + c0off;
    const unsigned short* c1p = lbuf + c1off;
    v16bf B0 = mk16(*(const v8bf*)c0p, *(const v8bf*)(c0p + 8));
    v16bf B1 = mk16(*(const v8bf*)c1p, *(const v8bf*)(c1p + 8));

    acc0 = wmma_bf16(A, B0, acc0);
    acc1 = wmma_bf16(A, B1, acc1);
    __syncthreads();
  };

  // phase 1: K from x_t  (no per-iteration pointer select)
#pragma unroll 4
  for (int kt = 0; kt < KTX; ++kt) kstep(xrow + kt * 32, kt);
  // phase 2: K from h_{t-1}
#pragma unroll 4
  for (int kt = KTX; kt < KT; ++kt) kstep(hrow + (kt - KTX) * 32, kt);

  // C/D layout: VGPR r, lane: M = m0 + 8*kh + r, N = nbase (+16 for acc1)
  float* op = preact + (size_t)(m0 + kh * 8) * G4H + nbase;
#pragma unroll
  for (int r = 0; r < 8; ++r) {
    op[(size_t)r * G4H]      = acc0[r];
    op[(size_t)r * G4H + 16] = acc1[r];
  }
}

// ---------------- per-step: LSTM cell elementwise ----------------
__global__ __launch_bounds__(256) void cell_kernel(const float* __restrict__ preact,
                                                   float* __restrict__ c,
                                                   unsigned short* __restrict__ hbf) {
  int i = blockIdx.x * blockDim.x + threadIdx.x;   // 65536
  int b = i >> 10;
  int n = i & 1023;
  const float* p = preact + (size_t)b * G4H;
  float f = sigm(p[n]);
  float ig = sigm(p[HSZ + n]);
  float o = sigm(p[2 * HSZ + n]);
  float g = tanh_fast(p[3 * HSZ + n]);
  float cn = f * c[i] + ig * g;
  c[i] = cn;
  hbf[i] = f2bf(o * tanh_fast(cn));
}

// ---------------- per-step: output GEMM [64 x 1024] x [1024 x 512] ----------------
__global__ __launch_bounds__(256) void out_wmma(
    const unsigned short* __restrict__ hbf,    // [B,HSZ]
    const unsigned short* __restrict__ wybf,   // [OSZ,HSZ]
    const float* __restrict__ by,              // [OSZ]
    float* __restrict__ out,                   // [S,B,OSZ]
    int t) {
  const int tid = threadIdx.x;
  const int wave = tid >> 5;
  const int lane = tid & 31;
  const int lr = lane & 15, kh = lane >> 4;
  const int m0 = (wave & 3) * 16;
  const int nhalf = wave >> 2;
  const int nblk = blockIdx.x * 64;
  const int nbase = nblk + nhalf * 32 + lr;

  const float b0 = by[nbase], b1 = by[nbase + 16];
  v8f acc0, acc1;
#pragma unroll
  for (int i = 0; i < 8; ++i) { acc0[i] = b0; acc1[i] = b1; }

  const unsigned short* arow = hbf + (size_t)(m0 + lr) * HSZ;
  const unsigned short* b0row = wybf + (size_t)nbase * HSZ + kh * 16;
  const unsigned short* b1row = wybf + (size_t)(nbase + 16) * HSZ + kh * 16;

#pragma unroll 4
  for (int kk = 0; kk < HSZ; kk += 32) {
    v16bf A = mk16(*(const v8bf*)(arow + kk + kh * 16),
                   *(const v8bf*)(arow + kk + kh * 16 + 8));
    v16bf B0 = mk16(*(const v8bf*)(b0row + kk), *(const v8bf*)(b0row + kk + 8));
    v16bf B1 = mk16(*(const v8bf*)(b1row + kk), *(const v8bf*)(b1row + kk + 8));
    acc0 = wmma_bf16(A, B0, acc0);
    acc1 = wmma_bf16(A, B1, acc1);
  }

  float* op = out + ((size_t)t * BATCH + m0 + kh * 8) * OSZ + nbase;
#pragma unroll
  for (int r = 0; r < 8; ++r) {
    op[(size_t)r * OSZ]      = acc0[r];
    op[(size_t)r * OSZ + 16] = acc1[r];
  }
}

// ---------------- host side ----------------
extern "C" void kernel_launch(void* const* d_in, const int* in_sizes, int n_in,
                              void* d_out, int out_size, void* d_ws, size_t ws_size,
                              hipStream_t stream) {
  (void)in_sizes; (void)n_in; (void)out_size; (void)ws_size;
  const float* x   = (const float*)d_in[0];
  const float* Wf  = (const float*)d_in[1];
  const float* bf_ = (const float*)d_in[2];
  const float* Wi  = (const float*)d_in[3];
  const float* bi_ = (const float*)d_in[4];
  const float* Wo  = (const float*)d_in[5];
  const float* bo_ = (const float*)d_in[6];
  const float* Wg  = (const float*)d_in[7];
  const float* bg_ = (const float*)d_in[8];
  const float* Wy  = (const float*)d_in[9];
  const float* by  = (const float*)d_in[10];

  char* ws = (char*)d_ws;
  size_t off = 0;
  auto take = [&](size_t bytes) {
    void* p = ws + off;
    off = (off + bytes + 255) & ~(size_t)255;
    return p;
  };
  unsigned short* wbf    = (unsigned short*)take((size_t)G4H * CSZ * 2);   // 12.6 MB
  unsigned short* wybf   = (unsigned short*)take((size_t)OSZ * HSZ * 2);   //  1.0 MB
  float*          ball   = (float*)take((size_t)G4H * 4);
  unsigned short* xbf    = (unsigned short*)take((size_t)S_LEN * BATCH * ISZ * 2); // 32 MB
  unsigned short* hbf    = (unsigned short*)take((size_t)BATCH * HSZ * 2);
  float*          cstate = (float*)take((size_t)BATCH * HSZ * 4);
  float*          preact = (float*)take((size_t)BATCH * G4H * 4);

  // one-time conversions (6,819,840 elements == 26640 * 256)
  prep_weights<<<26640, 256, 0, stream>>>(Wf, Wi, Wo, Wg, bf_, bi_, bo_, bg_, Wy,
                                          wbf, wybf, ball);
  prep_x<<<16384, 256, 0, stream>>>(x, xbf);          // 16M elems / 4 per thread
  init_state<<<256, 256, 0, stream>>>(hbf, cstate);

  for (int t = 0; t < S_LEN; ++t) {
    gates_wmma<<<64, 256, 0, stream>>>(xbf, hbf, wbf, ball, preact, t);
    cell_kernel<<<256, 256, 0, stream>>>(preact, cstate, hbf);
    out_wmma<<<8, 256, 0, stream>>>(hbf, wybf, by, (float*)d_out, t);
  }
}
